// AttentionMM_41257455846091
// MI455X (gfx1250) — matrix operand order repaired and checked by reading, not compiled, not run on hardware
//
#include <hip/hip_runtime.h>

typedef __attribute__((ext_vector_type(16))) _Float16 v16h;
typedef __attribute__((ext_vector_type(8)))  float    v8f;

constexpr int B_ = 32, T_ = 512, E_ = 512;
constexpr int G_ = 8;                 // m-groups for the final (mean) kernel

#define WMMA_F16(a,b,c) __builtin_amdgcn_wmma_f32_16x16x32_f16(false,(a),false,(b),(short)0,(c),false,false)

// ---- fragment loaders (CDNA5 wave32 WMMA 16x16x32 f16 layouts) ----
// A (16x32, MxK): lane half h, row M = lane&15.
//   elem i -> K = 16h + i
__device__ __forceinline__ v16h load_a_f16(const _Float16* __restrict__ A, int row, int k0, int ld, int h) {
  v16h a;
  const _Float16* p = A + (size_t)row * ld + k0 + 16 * h;
#pragma unroll
  for (int i = 0; i < 8; ++i) a[i] = p[i];
#pragma unroll
  for (int i = 0; i < 8; ++i) a[8 + i] = p[8 + i];
  return a;
}
__device__ __forceinline__ v16h load_a_f32(const float* __restrict__ A, int row, int k0, int ld, int h) {
  v16h a;
  const float* p = A + (size_t)row * ld + k0 + 16 * h;
#pragma unroll
  for (int i = 0; i < 8; ++i) a[i] = (_Float16)p[i];
#pragma unroll
  for (int i = 0; i < 8; ++i) a[8 + i] = (_Float16)p[8 + i];
  return a;
}
// B (32x16, KxN): lane half h, col N = lane&15, elem i -> K = 16h + i.
// Source is K-major: BT[n][k] row-major with leading dim ld.
__device__ __forceinline__ v16h load_bt_f16(const _Float16* __restrict__ BT, int n, int k0, int ld, int h) {
  v16h b;
  const _Float16* p = BT + (size_t)n * ld + k0 + 16 * h;
#pragma unroll
  for (int i = 0; i < 16; ++i) b[i] = p[i];
  return b;
}
__device__ __forceinline__ v16h load_bt_f32(const float* __restrict__ BT, int n, int k0, int ld, int h) {
  v16h b;
  const float* p = BT + (size_t)n * ld + k0 + 16 * h;
#pragma unroll
  for (int i = 0; i < 16; ++i) b[i] = (_Float16)p[i];
  return b;
}

// ---- transpose + fp32->f16 convert: src [R][C] f32 -> dst [C][R] f16 ----
__global__ void k_trans_f32(const float* __restrict__ src, _Float16* __restrict__ dst,
                            int R, int C, long long sStride, long long dStride) {
  __shared__ _Float16 t[32][33];
  long long b = blockIdx.z;
  const float* s = src + b * sStride;
  _Float16* d = dst + b * dStride;
  int c0 = blockIdx.x * 32, r0 = blockIdx.y * 32;
  int tx = threadIdx.x, ty = threadIdx.y;
#pragma unroll
  for (int i = 0; i < 32; i += 8) t[ty + i][tx] = (_Float16)s[(size_t)(r0 + ty + i) * C + (c0 + tx)];
  __syncthreads();
#pragma unroll
  for (int i = 0; i < 32; i += 8) d[(size_t)(c0 + ty + i) * R + (r0 + tx)] = t[tx][ty + i];
}

// ---- f16 transpose with explicit leading dims (alpha -> alphaT inside score buffer) ----
__global__ void k_trans_f16(const _Float16* __restrict__ src, int ldS, long long sStride,
                            _Float16* __restrict__ dst, int ldD, long long dStride) {
  __shared__ _Float16 t[32][33];
  long long b = blockIdx.z;
  const _Float16* s = src + b * sStride;
  _Float16* d = dst + b * dStride;
  int c0 = blockIdx.x * 32, r0 = blockIdx.y * 32;
  int tx = threadIdx.x, ty = threadIdx.y;
#pragma unroll
  for (int i = 0; i < 32; i += 8) t[ty + i][tx] = s[(size_t)(r0 + ty + i) * ldS + (c0 + tx)];
  __syncthreads();
#pragma unroll
  for (int i = 0; i < 32; i += 8) d[(size_t)(c0 + ty + i) * ldD + (r0 + tx)] = t[tx][ty + i];
}

// ---- scores: S[b] = x1[b] (TxE) . x2[b]^T -> fp32 [T][T]; 32x64 tile per wave ----
__global__ void k_scores(const float* __restrict__ x1, const float* __restrict__ x2,
                         float* __restrict__ S) {
  int wave = blockIdx.x * (blockDim.x >> 5) + (threadIdx.x >> 5);
  int lane = threadIdx.x & 31;
  int h = lane >> 4, lo = lane & 15;
  const int mt = T_ / 32, nt = T_ / 64;   // 16 x 8 tiles per batch
  int b = wave / (mt * nt);
  int rem = wave % (mt * nt);
  int mi = rem / nt, ni = rem % nt;
  int row0 = mi * 32, col0 = ni * 64;
  const float* X1 = x1 + (size_t)b * T_ * E_;
  const float* X2 = x2 + (size_t)b * T_ * E_;
  v8f c[8] = {};
  for (int k0 = 0; k0 < E_; k0 += 32) {
    v16h a0 = load_a_f32(X1, row0 + lo,      k0, E_, h);
    v16h a1 = load_a_f32(X1, row0 + 16 + lo, k0, E_, h);
#pragma unroll
    for (int n = 0; n < 4; ++n) {
      v16h bb = load_bt_f32(X2, col0 + n * 16 + lo, k0, E_, h);  // B = x2^T: K-major == x2 rows
      c[2 * n]     = WMMA_F16(a0, bb, c[2 * n]);
      c[2 * n + 1] = WMMA_F16(a1, bb, c[2 * n + 1]);
    }
  }
  float* Sp = S + (size_t)b * T_ * T_ + (size_t)row0 * T_ + col0;
#pragma unroll
  for (int n = 0; n < 4; ++n)
#pragma unroll
    for (int r = 0; r < 8; ++r) {
      Sp[(size_t)(r + 8 * h) * T_      + n * 16 + lo] = c[2 * n][r];
      Sp[(size_t)(16 + r + 8 * h) * T_ + n * 16 + lo] = c[2 * n + 1][r];
    }
}

// ---- softmax over rows; writes alpha as f16 into low half of each fp32 row ----
__global__ void k_softmax(const float* __restrict__ S, _Float16* __restrict__ alpha, int ldA) {
  int row = blockIdx.x;  // b*T + t
  const float* s = S + (size_t)row * T_;
  _Float16* a = alpha + (size_t)row * ldA;
  __shared__ float red[256];
  int tid = threadIdx.x;
  float v0 = s[tid], v1 = s[tid + 256];
  red[tid] = fmaxf(v0, v1);
  __syncthreads();
  for (int st = 128; st > 0; st >>= 1) { if (tid < st) red[tid] = fmaxf(red[tid], red[tid + st]); __syncthreads(); }
  float m = red[0];
  __syncthreads();
  float e0 = __expf(v0 - m), e1 = __expf(v1 - m);
  red[tid] = e0 + e1;
  __syncthreads();
  for (int st = 128; st > 0; st >>= 1) { if (tid < st) red[tid] += red[tid + st]; __syncthreads(); }
  float inv = 1.0f / red[0];
  a[tid] = (_Float16)(e0 * inv);
  a[tid + 256] = (_Float16)(e1 * inv);
}

// ---- f16 GEMM: O[MxN] = A[MxK] . BT[NxK]^T, 32x64 tile per wave, batched ----
__global__ void k_gemm_f16(const _Float16* __restrict__ A, int ldA, long long strA,
                           const _Float16* __restrict__ BT, int ldB, long long strB,
                           _Float16* __restrict__ O, int ldO, long long strO,
                           int M, int N, int K) {
  int wave = blockIdx.x * (blockDim.x >> 5) + (threadIdx.x >> 5);
  int lane = threadIdx.x & 31;
  int h = lane >> 4, lo = lane & 15;
  int mt = M / 32, nt = N / 64, tpb = mt * nt;
  int b = wave / tpb;
  int rem = wave % tpb;
  int mi = rem / nt, ni = rem % nt;
  int row0 = mi * 32, col0 = ni * 64;
  const _Float16* Ab = A + (long long)b * strA;
  const _Float16* Bb = BT + (long long)b * strB;
  v8f c[8] = {};
  for (int k0 = 0; k0 < K; k0 += 32) {
    v16h a0 = load_a_f16(Ab, row0 + lo,      k0, ldA, h);
    v16h a1 = load_a_f16(Ab, row0 + 16 + lo, k0, ldA, h);
#pragma unroll
    for (int n = 0; n < 4; ++n) {
      v16h bb = load_bt_f16(Bb, col0 + n * 16 + lo, k0, ldB, h);
      c[2 * n]     = WMMA_F16(a0, bb, c[2 * n]);
      c[2 * n + 1] = WMMA_F16(a1, bb, c[2 * n + 1]);
    }
  }
  _Float16* Ob = O + (long long)b * strO + (long long)row0 * ldO + col0;
#pragma unroll
  for (int n = 0; n < 4; ++n)
#pragma unroll
    for (int r = 0; r < 8; ++r) {
      Ob[(long long)(r + 8 * h) * ldO      + n * 16 + lo] = (_Float16)c[2 * n][r];
      Ob[(long long)(16 + r + 8 * h) * ldO + n * 16 + lo] = (_Float16)c[2 * n + 1][r];
    }
}

// ---- final partials: part[g][b][col] = sum over rows in group g of tanh(X.U^T + A2.V^T) ----
__global__ void k_out(const float* __restrict__ X, const _Float16* __restrict__ UT,
                      const _Float16* __restrict__ A2, const _Float16* __restrict__ VT,
                      float* __restrict__ part) {
  int wave = blockIdx.x * (blockDim.x >> 5) + (threadIdx.x >> 5);
  int lane = threadIdx.x & 31;
  int h = lane >> 4, lo = lane & 15;
  const int ntiles = E_ / 16;
  int g = wave % G_;
  int rem = wave / G_;
  int b = rem / ntiles, ni = rem % ntiles;
  const float* Xb = X + (size_t)b * T_ * E_;
  const _Float16* Ab = A2 + (size_t)b * T_ * E_;
  int col = ni * 16 + lo;
  const int mPerG = (T_ / 16) / G_;    // 4 m-tiles per group
  float acc = 0.0f;
  for (int mm = 0; mm < mPerG; ++mm) {
    int row = (g * mPerG + mm) * 16 + lo;
    v8f c = {};
    for (int k0 = 0; k0 < E_; k0 += 32) {
      v16h a  = load_a_f32(Xb, row, k0, E_, h);
      v16h bb = load_bt_f16(UT, col, k0, E_, h);
      c = WMMA_F16(a, bb, c);
    }
    for (int k0 = 0; k0 < E_; k0 += 32) {
      v16h a  = load_a_f16(Ab, row, k0, E_, h);
      v16h bb = load_bt_f16(VT, col, k0, E_, h);
      c = WMMA_F16(a, bb, c);
    }
#pragma unroll
    for (int r = 0; r < 8; ++r) acc += tanhf(c[r]);
  }
  acc += __shfl_xor(acc, 16, 32);  // lanes lo and lo+16 hold the same output column
  if (h == 0) part[((size_t)g * B_ + b) * E_ + col] = acc;
}

// ---- deterministic reduction of the G_ partials, applies the 1/T mean, fills d_out ----
__global__ void k_reduce(const float* __restrict__ p1, const float* __restrict__ p2,
                         float* __restrict__ out) {
  int i = blockIdx.x * blockDim.x + threadIdx.x;  // 0 .. B*E-1
  int b = i / E_, e = i % E_;
  float s1 = 0.0f, s2 = 0.0f;
#pragma unroll
  for (int g = 0; g < G_; ++g) {
    s1 += p1[((size_t)g * B_ + b) * E_ + e];
    s2 += p2[((size_t)g * B_ + b) * E_ + e];
  }
  out[(size_t)b * (2 * E_) + e]      = s1 * (1.0f / T_);
  out[(size_t)b * (2 * E_) + E_ + e] = s2 * (1.0f / T_);
}

extern "C" void kernel_launch(void* const* d_in, const int* in_sizes, int n_in,
                              void* d_out, int out_size, void* d_ws, size_t ws_size,
                              hipStream_t stream) {
  (void)in_sizes; (void)n_in; (void)out_size; (void)ws_size;
  const float* x1 = (const float*)d_in[0];
  const float* x2 = (const float*)d_in[1];
  const float* U1 = (const float*)d_in[2];
  const float* U2 = (const float*)d_in[3];
  const float* V1 = (const float*)d_in[4];
  const float* V2 = (const float*)d_in[5];
  float* out = (float*)d_out;

  // workspace layout (~99 MB, L2-resident on MI455X)
  char* ws = (char*)d_ws;
  size_t off = 0;
  float* S = (float*)(ws + off);          off += (size_t)B_ * T_ * T_ * sizeof(float);   // 32 MB
  _Float16* x1T = (_Float16*)(ws + off);  off += (size_t)B_ * E_ * T_ * 2;               // 16 MB
  _Float16* x2T = (_Float16*)(ws + off);  off += (size_t)B_ * E_ * T_ * 2;               // 16 MB
  _Float16* a1t = (_Float16*)(ws + off);  off += (size_t)B_ * T_ * E_ * 2;               // 16 MB
  _Float16* a2t = (_Float16*)(ws + off);  off += (size_t)B_ * T_ * E_ * 2;               // 16 MB
  _Float16* U1T = (_Float16*)(ws + off);  off += (size_t)E_ * E_ * 2;
  _Float16* U2T = (_Float16*)(ws + off);  off += (size_t)E_ * E_ * 2;
  _Float16* V1T = (_Float16*)(ws + off);  off += (size_t)E_ * E_ * 2;
  _Float16* V2T = (_Float16*)(ws + off);  off += (size_t)E_ * E_ * 2;
  float* part1 = (float*)(ws + off);      off += (size_t)G_ * B_ * E_ * sizeof(float);   // 0.5 MB
  float* part2 = (float*)(ws + off);      off += (size_t)G_ * B_ * E_ * sizeof(float);   // 0.5 MB

  // alpha (f16) lives in the low half of each score row; alphaT in the high half
  _Float16* alpha  = (_Float16*)S;
  _Float16* alphaT = (_Float16*)S + T_;
  const int ldAl = 2 * T_;
  const long long strAl = (long long)2 * T_ * T_;

  dim3 tb(32, 8, 1);
  // transposed f16 operand copies
  k_trans_f32<<<dim3(E_ / 32, T_ / 32, B_), tb, 0, stream>>>(x1, x1T, T_, E_, (long long)T_ * E_, (long long)E_ * T_);
  k_trans_f32<<<dim3(E_ / 32, T_ / 32, B_), tb, 0, stream>>>(x2, x2T, T_, E_, (long long)T_ * E_, (long long)E_ * T_);
  k_trans_f32<<<dim3(E_ / 32, E_ / 32, 1), tb, 0, stream>>>(U1, U1T, E_, E_, 0, 0);
  k_trans_f32<<<dim3(E_ / 32, E_ / 32, 1), tb, 0, stream>>>(U2, U2T, E_, E_, 0, 0);
  k_trans_f32<<<dim3(E_ / 32, E_ / 32, 1), tb, 0, stream>>>(V1, V1T, E_, E_, 0, 0);
  k_trans_f32<<<dim3(E_ / 32, E_ / 32, 1), tb, 0, stream>>>(V2, V2T, E_, E_, 0, 0);

  // 32x64 tiles: 16*8 tiles per batch * 32 batches = 4096 waves, 8 waves/block
  const int gemmWaves = B_ * (T_ / 32) * (T_ / 64);
  k_scores<<<gemmWaves / 8, 256, 0, stream>>>(x1, x2, S);
  k_softmax<<<B_ * T_, 256, 0, stream>>>(S, alpha, ldAl);
  k_trans_f16<<<dim3(T_ / 32, T_ / 32, B_), tb, 0, stream>>>(alpha, ldAl, strAl, alphaT, ldAl, strAl);

  // a2t = alpha . x1   (B from x1^T);  a1t = alpha^T . x2  (B from x2^T)
  k_gemm_f16<<<gemmWaves / 8, 256, 0, stream>>>(alpha,  ldAl, strAl, x1T, T_, (long long)E_ * T_,
                                                a2t, E_, (long long)T_ * E_, T_, E_, T_);
  k_gemm_f16<<<gemmWaves / 8, 256, 0, stream>>>(alphaT, ldAl, strAl, x2T, T_, (long long)E_ * T_,
                                                a1t, E_, (long long)T_ * E_, T_, E_, T_);

  // final GEMMs + tanh + row-group partial means (8192 waves each), then deterministic reduce
  const int wavesOut = B_ * (E_ / 16) * G_;
  k_out<<<wavesOut / 8, 256, 0, stream>>>(x1, U1T, a1t, V1T, part1);
  k_out<<<wavesOut / 8, 256, 0, stream>>>(x2, U2T, a2t, V2T, part2);
  k_reduce<<<(B_ * E_) / 256, 256, 0, stream>>>(part1, part2, out);
}
